// RWKV_DiffusionDeCoderLayer_78391743086608
// MI455X (gfx1250) — matrix operand order repaired and checked by reading, not compiled, not run on hardware
//
#include <hip/hip_runtime.h>
#include <math.h>

// ---------------------------------------------------------------------------
// RWKV diffusion decoder layer for MI455X (gfx1250, wave32, WMMA).
// GEMMs: v_wmma_f32_16x16x32_bf16 with double-buffered LDS staging via
// GLOBAL_LOAD_ASYNC_TO_LDS_B128 (ASYNCcnt). RWKV7 scan keeps the 64x64 state
// in VGPRs (2 rows/lane) with wave32 shuffle broadcasts; LN/softmax/groupnorm
// are LDS reductions.
// ---------------------------------------------------------------------------

#define B_    64
#define M_    20
#define T_    8
#define C_    256
#define H_    4
#define N_    64
#define FFN_  1024
#define NA_   32
#define DIN_  280
#define PI_F  3.14159265358979323846f

typedef __attribute__((ext_vector_type(16))) __bf16 v16bf;
typedef __attribute__((ext_vector_type(8)))  float  v8f;

enum { ACT_NONE = 0, ACT_RELU = 1, ACT_SIGMOID = 2, ACT_TANH = 3, ACT_NEGEXP = 4 };

__device__ __forceinline__ __bf16 f2bf(float f) {
  union { float f; unsigned u; } x; x.f = f;
  unsigned r = x.u + 0x7FFFu + ((x.u >> 16) & 1u);   // round-to-nearest-even
  union { unsigned short s; __bf16 b; } y; y.s = (unsigned short)(r >> 16);
  return y.b;
}

// ---- async global->LDS staging (CDNA5). Probed; falls back to sync copy. ----
#if __has_builtin(__builtin_amdgcn_global_load_async_to_lds_b128) && \
    __has_builtin(__builtin_amdgcn_s_wait_asynccnt)
#define GEMM_ASYNC 1
typedef __attribute__((vector_size(16))) int vi4;
typedef vi4 __attribute__((address_space(1)))* g_vi4_ptr;
typedef vi4 __attribute__((address_space(3)))* l_vi4_ptr;
__device__ __forceinline__ void stage16(void* lds, const void* g) {
  __builtin_amdgcn_global_load_async_to_lds_b128(
      (g_vi4_ptr)(void*)g, (l_vi4_ptr)lds, 0, 0);
}
__device__ __forceinline__ void stage_join() { __builtin_amdgcn_s_wait_asynccnt(0); }
#else
__device__ __forceinline__ void stage16(void* lds, const void* g) {
  *(uint4*)lds = *(const uint4*)g;
}
__device__ __forceinline__ void stage_join() {}
#endif

// ---------------------------------------------------------------------------
// GEMM: Y[rows x N] = act(A[rows x K] * W[K x N] + bias)
// block = 128 threads / 4 waves; tile = 16 rows x 64 cols; K-step = 32.
// Assumes rows % 16 == 0 (true at every call site: 1280/3328/1344/64).
// EDGE=false (N % 64 == 0): fully branch-free async staging, double-buffered
// so the DMA for tile k+1 overlaps the WMMA of tile k.
// EDGE=true (ragged N: 6/24/32/80/720, all tiny): branchless clamped scalar
// staging for W. K tail (K=280 heads) is a zero-filled sync epilogue.
// ---------------------------------------------------------------------------
template <bool EDGE>
__global__ void __launch_bounds__(128) gemm_kernel(
    const float* __restrict__ A, const float* __restrict__ W,
    const float* __restrict__ bias, float* __restrict__ Y,
    int rows, int K, int N, int act) {
  __shared__ float sA[2][16 * 32];   // [buf][row][k]  2 x 2 KB
  __shared__ float sB[2][32 * 64];   // [buf][k][n]    2 x 8 KB
  const int tid  = threadIdx.x;
  const int wave = tid >> 5;
  const int lane = tid & 31;
  const int half = lane >> 4;
  const int l15  = lane & 15;
  const int nb0  = blockIdx.x * 64;
  const int m0   = blockIdx.y * 16;
  const int col  = nb0 + wave * 16 + l15;     // this lane's output column
  const int sa_row = tid >> 3;                // 0..15
  const int sa_k4  = (tid & 7) * 4;           // 0..28 step 4
  const int Kmain  = K & ~31;

  auto stage_tile = [&](int buf, int kt) {
    stage16(&sA[buf][sa_row * 32 + sa_k4],
            &A[(size_t)(m0 + sa_row) * K + kt + sa_k4]);
#pragma unroll
    for (int i = 0; i < 4; ++i) {
      int lin = i * 128 + tid;
      int kr  = lin >> 4;                     // 0..31
      int c4  = (lin & 15) * 4;               // 0..60 step 4
      if (!EDGE) {
        stage16(&sB[buf][kr * 64 + c4], &W[(size_t)(kt + kr) * N + nb0 + c4]);
      } else {
#pragma unroll
        for (int j = 0; j < 4; ++j) {
          int cj = nb0 + c4 + j;
          sB[buf][kr * 64 + c4 + j] = W[(size_t)(kt + kr) * N + (cj < N ? cj : N - 1)];
        }
      }
    }
  };
  auto compute = [&](int buf, v8f acc) -> v8f {
    v16bf av, bv;
#pragma unroll
    for (int e = 0; e < 16; ++e) {            // ISA 7.12.2 fragment layouts
      int ka = (e & 7) + ((e & 8) ? 16 : 0) + (half ? 8 : 0);
      int kb = (e & 7) + ((e & 8) ? 16 : 0) + (half ? 8 : 0);
      av[e] = f2bf(sA[buf][l15 * 32 + ka]);
      bv[e] = f2bf(sB[buf][kb * 64 + wave * 16 + l15]);
    }
    return __builtin_amdgcn_wmma_f32_16x16x32_bf16(false, av, false, bv,
                                                   (short)0, acc, false, false);
  };

  v8f acc = {};
  int cur = 0;
  if (Kmain > 0) {
    stage_tile(0, 0);                         // prologue: tile 0 in flight
    for (int k0 = 0; k0 < Kmain; k0 += 32) {
      stage_join();                           // my async copies for tile k0 done
      __syncthreads();                        // all waves' copies visible;
                                              // also: everyone finished tile k0-32
      if (k0 + 32 < Kmain) stage_tile(cur ^ 1, k0 + 32);  // overlap DMA w/ WMMA
      acc = compute(cur, acc);
      cur ^= 1;
    }
  }

  if (Kmain < K) {                            // zero-filled sync tail
    __syncthreads();
#pragma unroll
    for (int j = 0; j < 4; ++j) {
      int kk = Kmain + sa_k4 + j;
      sA[cur][sa_row * 32 + sa_k4 + j] =
          (kk < K) ? A[(size_t)(m0 + sa_row) * K + kk] : 0.f;
    }
#pragma unroll
    for (int i = 0; i < 4; ++i) {
      int lin = i * 128 + tid;
      int kr  = lin >> 4;
      int c4  = (lin & 15) * 4;
      int kk  = Kmain + kr;
#pragma unroll
      for (int j = 0; j < 4; ++j) {
        int cj = nb0 + c4 + j;
        sB[cur][kr * 64 + c4 + j] =
            (kk < K) ? W[(size_t)kk * N + (cj < N ? cj : N - 1)] : 0.f;
      }
    }
    __syncthreads();
    acc = compute(cur, acc);
  }

  if (col < N) {
    float bz = bias ? bias[col] : 0.f;
#pragma unroll
    for (int r = 0; r < 8; ++r) {
      int row = m0 + r + (half ? 8 : 0);      // C/D layout: VGPR r -> M = r + 8*half
      float v = acc[r] + bz;
      if      (act == ACT_RELU)    v = fmaxf(v, 0.f);
      else if (act == ACT_SIGMOID) v = 1.f / (1.f + __expf(-v));
      else if (act == ACT_TANH)    v = tanhf(v);
      else if (act == ACT_NEGEXP)  v = __expf(-__expf(v));
      Y[(size_t)row * N + col] = v;
    }
  }
}

// ---------------------------------------------------------------------------
// LayerNorm over C=256 (optional residual R): Y = LN(X [+ R]) * g + b
// ---------------------------------------------------------------------------
__global__ void ln_kernel(const float* __restrict__ X, const float* __restrict__ R,
                          const float* __restrict__ gg, const float* __restrict__ bb,
                          float* __restrict__ Y, int rows) {
  int row = blockIdx.x, c = threadIdx.x;
  if (row >= rows) return;
  size_t base = (size_t)row * C_;
  __shared__ float red[C_];
  float x = X[base + c] + (R ? R[base + c] : 0.f);
  red[c] = x; __syncthreads();
  for (int s = 128; s >= 1; s >>= 1) { if (c < s) red[c] += red[c + s]; __syncthreads(); }
  float mu = red[0] * (1.f / C_); __syncthreads();
  float d = x - mu;
  red[c] = d * d; __syncthreads();
  for (int s = 128; s >= 1; s >>= 1) { if (c < s) red[c] += red[c + s]; __syncthreads(); }
  float var = red[0] * (1.f / C_);
  Y[base + c] = d * rsqrtf(var + 1e-5f) * gg[c] + bb[c];
}

// ---------------------------------------------------------------------------
// Cross-BEV: bilinear gather of T=8 points + per-head softmax attention.
// ---------------------------------------------------------------------------
__global__ void cross_bev_kernel(const float* __restrict__ traj, const float* __restrict__ pts,
                                 const float* __restrict__ bev, const float* __restrict__ Wattn,
                                 const float* __restrict__ battn, float* __restrict__ outp,
                                 const int* __restrict__ pHb, const int* __restrict__ pWb) {
  int b = blockIdx.y, m = blockIdx.x, tid = threadIdx.x;
  int Hb = *pHb, Wb = *pWb;
  __shared__ float s_traj[C_];
  __shared__ float s_wts[H_ * T_];
  __shared__ int   s_i[T_][4];
  __shared__ float s_f[T_][4];
  s_traj[tid] = traj[((size_t)b * M_ + m) * C_ + tid];
  __syncthreads();
  if (tid < H_ * T_) {                 // logits (H,T)
    float acc = battn[tid];
    for (int c = 0; c < C_; ++c) acc += s_traj[c] * Wattn[c * (H_ * T_) + tid];
    s_wts[tid] = acc;
  }
  if (tid < T_) {                      // bilinear indices/weights
    int t = tid;
    float px = pts[(((size_t)b * M_ + m) * T_ + t) * 2 + 0];
    float py = pts[(((size_t)b * M_ + m) * T_ + t) * 2 + 1];
    float xp = (px + 1.f) * 0.5f * (float)Wb - 0.5f;
    float yp = (py + 1.f) * 0.5f * (float)Hb - 0.5f;
    float x0 = floorf(xp), y0 = floorf(yp);
    float wx = xp - x0, wy = yp - y0;
    int x0i = (int)x0; x0i = x0i < 0 ? 0 : (x0i > Wb - 1 ? Wb - 1 : x0i);
    int x1i = x0i + 1; x1i = x1i > Wb - 1 ? Wb - 1 : x1i;
    int y0i = (int)y0; y0i = y0i < 0 ? 0 : (y0i > Hb - 1 ? Hb - 1 : y0i);
    int y1i = y0i + 1; y1i = y1i > Hb - 1 ? Hb - 1 : y1i;
    s_i[t][0] = y0i * Wb + x0i; s_i[t][1] = y0i * Wb + x1i;
    s_i[t][2] = y1i * Wb + x0i; s_i[t][3] = y1i * Wb + x1i;
    s_f[t][0] = (1.f - wx) * (1.f - wy); s_f[t][1] = wx * (1.f - wy);
    s_f[t][2] = (1.f - wx) * wy;         s_f[t][3] = wx * wy;
  }
  __syncthreads();
  if (tid < H_) {                      // softmax over T per head
    float mx = -1e30f;
    for (int t = 0; t < T_; ++t) mx = fmaxf(mx, s_wts[tid * T_ + t]);
    float s = 0.f;
    for (int t = 0; t < T_; ++t) s += __expf(s_wts[tid * T_ + t] - mx);
    float inv = 1.f / s;
    for (int t = 0; t < T_; ++t) s_wts[tid * T_ + t] = __expf(s_wts[tid * T_ + t] - mx) * inv;
  }
  __syncthreads();
  const float* base = bev + (size_t)b * Hb * Wb * C_;
  int hh = tid >> 6;
  float acc = 0.f;
  for (int t = 0; t < T_; ++t) {
    float sv = s_f[t][0] * base[(size_t)s_i[t][0] * C_ + tid]
             + s_f[t][1] * base[(size_t)s_i[t][1] * C_ + tid]
             + s_f[t][2] * base[(size_t)s_i[t][2] * C_ + tid]
             + s_f[t][3] * base[(size_t)s_i[t][3] * C_ + tid];
    acc += s_wts[hh * T_ + t] * sv;
  }
  outp[((size_t)b * M_ + m) * C_ + tid] = acc;
}

// ------------------------- elementwise helpers ------------------------------
__global__ void concat_kernel(const float* __restrict__ ctx, const float* __restrict__ x,
                              float* __restrict__ seq, int Lc) {
  int L = Lc + M_;
  int i = blockIdx.x * 256 + threadIdx.x;
  int total = B_ * L * C_; if (i >= total) return;
  int c = i % C_, l = (i / C_) % L, b = i / (C_ * L);
  seq[i] = (l < Lc) ? ctx[((size_t)b * Lc + l) * C_ + c]
                    : x[((size_t)b * M_ + (l - Lc)) * C_ + c];
}

__global__ void slice_kernel(const float* __restrict__ seq, float* __restrict__ x, int Lc) {
  int L = Lc + M_;
  int i = blockIdx.x * 256 + threadIdx.x;
  int total = B_ * M_ * C_; if (i >= total) return;
  int c = i % C_, m = (i / C_) % M_, b = i / (C_ * M_);
  x[i] = seq[((size_t)b * L + Lc + m) * C_ + c];
}

__global__ void mix_kernel(const float* __restrict__ X, const float* __restrict__ coef,
                           float* __restrict__ Y, int L) {
  int i = blockIdx.x * 256 + threadIdx.x;
  int total = B_ * L * C_; if (i >= total) return;
  int c = i % C_, l = (i / C_) % L;
  float x = X[i];
  float xp = (l == 0) ? 0.f : X[i - C_];
  Y[i] = x + (xp - x) * coef[c];
}

__global__ void add_kernel(float* __restrict__ Y, const float* __restrict__ X, int total) {
  int i = blockIdx.x * 256 + threadIdx.x;
  if (i < total) Y[i] += X[i];
}

__global__ void kupdate_kernel(float* __restrict__ k, const float* __restrict__ a,
                               const float* __restrict__ k_a, int total) {
  int i = blockIdx.x * 256 + threadIdx.x;
  if (i >= total) return;
  k[i] *= (1.f + (a[i] - 1.f) * k_a[i % C_]);
}

// kk = (k_raw * k_k) / (||.||_head + 1e-6); one wave per (row, head)
__global__ void kknorm_kernel(const float* __restrict__ kraw, const float* __restrict__ k_k,
                              float* __restrict__ kk, int rowsH) {
  int gw = (blockIdx.x * blockDim.x + threadIdx.x) >> 5;
  int lane = threadIdx.x & 31;
  if (gw >= rowsH) return;
  int row = gw >> 2, hh = gw & 3;
  size_t base = (size_t)row * C_ + hh * N_;
  float a0 = kraw[base + lane]      * k_k[hh * N_ + lane];
  float a1 = kraw[base + 32 + lane] * k_k[hh * N_ + 32 + lane];
  float ss = a0 * a0 + a1 * a1;
#pragma unroll
  for (int off = 16; off >= 1; off >>= 1) ss += __shfl_down(ss, off, 32);
  ss = __shfl(ss, 0, 32);
  float inv = 1.f / (sqrtf(ss) + 1e-6f);
  kk[base + lane] = a0 * inv;
  kk[base + 32 + lane] = a1 * inv;
}

__global__ void vfirst_copy_kernel(const float* __restrict__ v, float* __restrict__ vf,
                                   float* __restrict__ vf_out, int L, int Lc) {
  int i = blockIdx.x * 256 + threadIdx.x;
  int total = B_ * M_ * C_; if (i >= total) return;
  int c = i % C_, m = (i / C_) % M_, b = i / (C_ * M_);
  float val = v[((size_t)b * L + Lc + m) * C_ + c];
  vf[i] = val; vf_out[i] = val;
}

__global__ void vmix_kernel(float* __restrict__ v, const float* __restrict__ vf,
                            const float* __restrict__ mix, int L, int Lc) {
  int i = blockIdx.x * 256 + threadIdx.x;
  int total = B_ * L * C_; if (i >= total) return;
  int c = i % C_, l = (i / C_) % L, b = i / (C_ * L);
  if (l < Lc) return;
  float vv = v[i];
  v[i] = vv + (vf[((size_t)b * M_ + (l - Lc)) * C_ + c] - vv) * mix[i];
}

// ---------------------------------------------------------------------------
// RWKV7 scan: one wave per (b, h); state S[64][64] register-resident,
// lane owns rows {lane, lane+32}; per-step vectors broadcast via shuffles.
// ---------------------------------------------------------------------------
__global__ void rwkv_scan_kernel(const float* __restrict__ r, const float* __restrict__ k,
                                 const float* __restrict__ v, const float* __restrict__ w,
                                 const float* __restrict__ kk, const float* __restrict__ a,
                                 float* __restrict__ y, int L) {
  int gw = (blockIdx.x * blockDim.x + threadIdx.x) >> 5;
  int lane = threadIdx.x & 31;
  if (gw >= B_ * H_) return;
  int b = gw >> 2, hh = gw & 3;
  float S0[N_], S1[N_];
#pragma unroll
  for (int j = 0; j < N_; ++j) { S0[j] = 0.f; S1[j] = 0.f; }
  for (int t = 0; t < L; ++t) {
    size_t base = ((size_t)b * L + t) * C_ + hh * N_;
    float r0 = r[base + lane],  r1 = r[base + 32 + lane];
    float k0 = k[base + lane],  k1 = k[base + 32 + lane];
    float v0 = v[base + lane],  v1 = v[base + 32 + lane];
    float w0 = w[base + lane],  w1 = w[base + 32 + lane];
    float q0 = kk[base + lane], q1 = kk[base + 32 + lane];
    float a0 = a[base + lane],  a1 = a[base + 32 + lane];
    float c0 = q0 * a0, c1 = q1 * a1;
    float sa0 = 0.f, sa1 = 0.f;
#pragma unroll
    for (int j = 0; j < 32; ++j) {
      float qj = __shfl(q0, j, 32);
      sa0 -= S0[j] * qj; sa1 -= S1[j] * qj;
    }
#pragma unroll
    for (int j = 0; j < 32; ++j) {
      float qj = __shfl(q1, j, 32);
      sa0 -= S0[32 + j] * qj; sa1 -= S1[32 + j] * qj;
    }
    float y0 = 0.f, y1 = 0.f;
#pragma unroll
    for (int j = 0; j < 32; ++j) {
      float wj = __shfl(w0, j, 32), kj = __shfl(k0, j, 32);
      float cj = __shfl(c0, j, 32), rj = __shfl(r0, j, 32);
      S0[j] = S0[j] * wj + sa0 * cj + v0 * kj;
      S1[j] = S1[j] * wj + sa1 * cj + v1 * kj;
      y0 += S0[j] * rj; y1 += S1[j] * rj;
    }
#pragma unroll
    for (int j = 0; j < 32; ++j) {
      float wj = __shfl(w1, j, 32), kj = __shfl(k1, j, 32);
      float cj = __shfl(c1, j, 32), rj = __shfl(r1, j, 32);
      S0[32 + j] = S0[32 + j] * wj + sa0 * cj + v0 * kj;
      S1[32 + j] = S1[32 + j] * wj + sa1 * cj + v1 * kj;
      y0 += S0[32 + j] * rj; y1 += S1[32 + j] * rj;
    }
    y[base + lane] = y0;
    y[base + 32 + lane] = y1;
  }
}

// group-norm(64) + lnx scale/bias + bonus + gate
__global__ void postscan_kernel(const float* __restrict__ y, const float* __restrict__ r,
                                const float* __restrict__ k, const float* __restrict__ v,
                                const float* __restrict__ g, const float* __restrict__ r_k,
                                const float* __restrict__ lnx_g, const float* __restrict__ lnx_b,
                                float* __restrict__ z, int rows) {
  int row = blockIdx.x, c = threadIdx.x;
  if (row >= rows) return;
  int hb = (c >> 6) << 6;
  size_t base = (size_t)row * C_;
  __shared__ float red[C_];
  float yv = y[base + c];
  red[c] = yv; __syncthreads();
  for (int s = 32; s >= 1; s >>= 1) { if ((c & 63) < s) red[c] += red[c + s]; __syncthreads(); }
  float mu = red[hb] * (1.f / N_); __syncthreads();
  float d = yv - mu;
  red[c] = d * d; __syncthreads();
  for (int s = 32; s >= 1; s >>= 1) { if ((c & 63) < s) red[c] += red[c + s]; __syncthreads(); }
  float var = red[hb] * (1.f / N_); __syncthreads();
  float yn = d * rsqrtf(var + 1e-5f) * lnx_g[c] + lnx_b[c];
  red[c] = r[base + c] * k[base + c] * r_k[c]; __syncthreads();
  for (int s = 32; s >= 1; s >>= 1) { if ((c & 63) < s) red[c] += red[c + s]; __syncthreads(); }
  float bsum = red[hb]; __syncthreads();
  z[base + c] = (yn + bsum * v[base + c]) * g[base + c];
}

__global__ void mish_kernel(const float* __restrict__ x, float* __restrict__ y, int total) {
  int i = blockIdx.x * 256 + threadIdx.x;
  if (i >= total) return;
  float v = x[i];
  float sp = (v > 20.f) ? v : log1pf(__expf(v));
  y[i] = v * tanhf(sp);
}

__global__ void modulate_kernel(float* __restrict__ x, const float* __restrict__ ss) {
  int i = blockIdx.x * 256 + threadIdx.x;
  int total = B_ * M_ * C_; if (i >= total) return;
  int c = i % C_, b = i / (M_ * C_);
  x[i] = x[i] * (1.f + ss[(size_t)b * 2 * C_ + c]) + ss[(size_t)b * 2 * C_ + C_ + c];
}

__global__ void poses_kernel(const float* __restrict__ reg, const float* __restrict__ pts,
                             float* __restrict__ out) {
  int i = blockIdx.x * 256 + threadIdx.x;
  int total = B_ * M_ * T_; if (i >= total) return;
  int t = i % T_; int bm = i / T_;
  float r0 = reg[(size_t)bm * (T_ * 3) + t * 3 + 0];
  float r1 = reg[(size_t)bm * (T_ * 3) + t * 3 + 1];
  float r2 = reg[(size_t)bm * (T_ * 3) + t * 3 + 2];
  float n0 = pts[((size_t)bm * T_ + t) * 2 + 0];
  float n1 = pts[((size_t)bm * T_ + t) * 2 + 1];
  out[(size_t)i * 3 + 0] = r0 + n0;
  out[(size_t)i * 3 + 1] = r1 + n1;
  out[(size_t)i * 3 + 2] = tanhf(r2) * PI_F;
}

__global__ void cat_kernel(const float* __restrict__ x, const float* __restrict__ d,
                           float* __restrict__ cat) {
  int i = blockIdx.x * 256 + threadIdx.x;
  int total = B_ * M_ * DIN_; if (i >= total) return;
  int c = i % DIN_, row = i / DIN_;
  cat[i] = (c < C_) ? x[(size_t)row * C_ + c] : d[(size_t)row * (T_ * 3) + (c - C_)];
}

// ------------------------------ host side -----------------------------------
static inline dim3 ewg(int n) { return dim3((n + 255) / 256); }

static void gemm(const float* A, const float* W, const float* bias, float* Y,
                 int rows, int K, int N, int act, hipStream_t s) {
  dim3 g((N + 63) / 64, (rows + 15) / 16);
  if ((N & 63) == 0)
    gemm_kernel<false><<<g, 128, 0, s>>>(A, W, bias, Y, rows, K, N, act);
  else
    gemm_kernel<true><<<g, 128, 0, s>>>(A, W, bias, Y, rows, K, N, act);
}

// rwkv param leaf offsets (sorted-key pytree order within attn{0,1})
// 0 Wk 1 Wo 2 Wr 3 Wv 4 a0 5 a1 6 a2 7 g1 8 g2 9 k_a 10 k_k 11 ln1_b 12 ln1_g
// 13 lnx_b 14 lnx_g 15 r_k 16 v0 17 v1 18 v2 19 w0 20 w1 21 w2
// 22 x_a 23 x_g 24 x_k 25 x_r 26 x_v 27 x_w
static void rwkv_block(void* const* d_in, int base, const float* ctx, int Lc,
                       float* b_seq, float* b_h, float* b_mx, float* b_r, float* b_k,
                       float* b_v, float* b_w, float* b_a, float* b_g, float* b_kk,
                       float* b_y, float* b_mid, float* b_z, float* b_at, float* b_mixs,
                       float* b_x, float* b_vf, float* vf_out, bool isFirst, hipStream_t s) {
  const int L = Lc + M_, rows = B_ * L, total = rows * C_;
  auto P = [&](int off) { return (const float*)d_in[base + off]; };

  concat_kernel<<<ewg(total), 256, 0, s>>>(ctx, b_x, b_seq, Lc);
  ln_kernel<<<rows, 256, 0, s>>>(b_seq, nullptr, P(12), P(11), b_h, rows);

  mix_kernel<<<ewg(total), 256, 0, s>>>(b_h, P(25), b_mx, L);          // xr
  gemm(b_mx, P(2), nullptr, b_r, rows, C_, C_, ACT_NONE, s);           // r
  mix_kernel<<<ewg(total), 256, 0, s>>>(b_h, P(24), b_mx, L);          // xk
  gemm(b_mx, P(0), nullptr, b_k, rows, C_, C_, ACT_NONE, s);           // k
  mix_kernel<<<ewg(total), 256, 0, s>>>(b_h, P(26), b_mx, L);          // xv
  gemm(b_mx, P(3), nullptr, b_v, rows, C_, C_, ACT_NONE, s);           // v
  if (!isFirst) {                                                       // v-residual mix
    gemm(b_mx, P(17), nullptr, b_mid, rows, C_, 32, ACT_NONE, s);
    gemm(b_mid, P(18), P(16), b_mixs, rows, 32, C_, ACT_SIGMOID, s);
  }
  mix_kernel<<<ewg(total), 256, 0, s>>>(b_h, P(27), b_mx, L);          // xw
  gemm(b_mx, P(20), nullptr, b_mid, rows, C_, 64, ACT_TANH, s);
  gemm(b_mid, P(21), P(19), b_w, rows, 64, C_, ACT_NEGEXP, s);         // decay
  mix_kernel<<<ewg(total), 256, 0, s>>>(b_h, P(22), b_mx, L);          // xa
  gemm(b_mx, P(5), nullptr, b_mid, rows, C_, 64, ACT_NONE, s);
  gemm(b_mid, P(6), P(4), b_a, rows, 64, C_, ACT_SIGMOID, s);          // a
  mix_kernel<<<ewg(total), 256, 0, s>>>(b_h, P(23), b_mx, L);          // xg
  gemm(b_mx, P(7), nullptr, b_mid, rows, C_, 128, ACT_SIGMOID, s);
  gemm(b_mid, P(8), nullptr, b_g, rows, 128, C_, ACT_NONE, s);         // g

  kknorm_kernel<<<(rows * H_ + 3) / 4, 128, 0, s>>>(b_k, P(10), b_kk, rows * H_);
  kupdate_kernel<<<ewg(total), 256, 0, s>>>(b_k, b_a, P(9), total);
  if (isFirst)
    vfirst_copy_kernel<<<ewg(B_ * M_ * C_), 256, 0, s>>>(b_v, b_vf, vf_out, L, Lc);
  else
    vmix_kernel<<<ewg(total), 256, 0, s>>>(b_v, b_vf, b_mixs, L, Lc);

  rwkv_scan_kernel<<<(B_ * H_ + 3) / 4, 128, 0, s>>>(b_r, b_k, b_v, b_w, b_kk, b_a, b_y, L);
  postscan_kernel<<<rows, 256, 0, s>>>(b_y, b_r, b_k, b_v, b_g, P(15), P(14), P(13), b_z, rows);
  gemm(b_z, P(1), nullptr, b_at, rows, C_, C_, ACT_NONE, s);           // @Wo
  add_kernel<<<ewg(total), 256, 0, s>>>(b_seq, b_at, total);
  slice_kernel<<<ewg(B_ * M_ * C_), 256, 0, s>>>(b_seq, b_x, Lc);
}

static void run_head(const float* cat, void* const* d_in, int base, float* outp, int dout,
                     float* hh, float* hh2, hipStream_t s) {
  auto P = [&](int off) { return (const float*)d_in[base + off]; };
  // head leaves (sorted): 0 b1 1 b2 2 b3 3 bb1 4 bb2 5 g1 6 g2 7 w1 8 w2 9 w3
  const int rows = B_ * M_;
  gemm(cat, P(7), P(0), hh, rows, DIN_, C_, ACT_RELU, s);
  ln_kernel<<<rows, 256, 0, s>>>(hh, nullptr, P(5), P(3), hh, rows);
  gemm(hh, P(8), P(1), hh2, rows, C_, C_, ACT_RELU, s);
  ln_kernel<<<rows, 256, 0, s>>>(hh2, nullptr, P(6), P(4), hh2, rows);
  gemm(hh2, P(9), P(2), outp, rows, C_, dout, ACT_NONE, s);
}

extern "C" void kernel_launch(void* const* d_in, const int* in_sizes, int n_in,
                              void* d_out, int out_size, void* d_ws, size_t ws_size,
                              hipStream_t stream) {
  (void)in_sizes; (void)n_in; (void)out_size; (void)ws_size;
  // ---- inputs (setup_inputs dict order; params pytree-flattened at 6..115) ----
  const float* traj   = (const float*)d_in[0];
  const float* pts    = (const float*)d_in[1];
  const float* bev    = (const float*)d_in[2];
  const float* agents = (const float*)d_in[3];
  const float* ego    = (const float*)d_in[4];
  const float* te     = (const float*)d_in[5];
  const int AREA0 = 6, ATTN0 = 16, ATTN1 = 44, CBEV0 = 72, COL0 = 78, SCORE0 = 106;
  auto P = [&](int i) { return (const float*)d_in[i]; };
  const int* pHb = (const int*)d_in[116];
  const int* pWb = (const int*)d_in[117];
  float* out = (float*)d_out;
  const size_t O_POSES = 0, O_SCORE = 30720, O_COL = 38400, O_AREA = 960000, O_VF = 1062400;

  // ---- workspace carve-up ----
  const size_t SEQ = (size_t)B_ * 52 * C_;         // 851968 floats
  const size_t XMC = (size_t)B_ * M_ * C_;         // 327680 floats
  float* ws = (float*)d_ws;
  float* b_seq  = ws;
  float* b_h    = ws + 1 * SEQ;
  float* b_mx   = ws + 2 * SEQ;
  float* b_r    = ws + 3 * SEQ;
  float* b_k    = ws + 4 * SEQ;
  float* b_v    = ws + 5 * SEQ;
  float* b_w    = ws + 6 * SEQ;
  float* b_a    = ws + 7 * SEQ;
  float* b_g    = ws + 8 * SEQ;
  float* b_kk   = ws + 9 * SEQ;
  float* b_y    = ws + 10 * SEQ;
  float* b_mid  = ws + 11 * SEQ;
  float* b_z    = ws + 12 * SEQ;
  float* b_at   = ws + 13 * SEQ;
  float* b_mixs = ws + 14 * SEQ;
  float* b_x    = ws + 15 * SEQ;
  float* b_x2   = b_x   + XMC;
  float* b_vf   = b_x2  + XMC;
  float* b_ffn  = b_vf  + XMC;                     // 1280x1024
  float* b_hh   = b_ffn + (size_t)B_ * M_ * FFN_;
  float* b_hh2  = b_hh  + XMC;
  float* b_reg  = b_hh2 + XMC;                     // 1280x24
  float* b_ss   = b_reg + (size_t)B_ * M_ * T_ * 3;// 64x512
  float* b_te   = b_ss  + (size_t)B_ * 2 * C_;     // 64x256
  float* b_cat  = b_te  + (size_t)B_ * C_;         // 1280x280
  float* b_cb   = b_cat + (size_t)B_ * M_ * DIN_;  // 1280x256 cross-bev attn out

  // ---- 1) cross-BEV attention + Wo + residual LN ----
  cross_bev_kernel<<<dim3(M_, B_), 256, 0, stream>>>(
      traj, pts, bev, P(CBEV0 + 0), P(CBEV0 + 2), b_cb, pHb, pWb);
  gemm(b_cb, P(CBEV0 + 1), P(CBEV0 + 3), b_x2, B_ * M_, C_, C_, ACT_NONE, stream);
  ln_kernel<<<B_ * M_, 256, 0, stream>>>(b_x2, traj, P(CBEV0 + 5), P(CBEV0 + 4), b_x, B_ * M_);

  // ---- 2) RWKV block 0 (ctx = agents, Lc = 32, is_first) + LN n1 ----
  rwkv_block(d_in, ATTN0, agents, NA_, b_seq, b_h, b_mx, b_r, b_k, b_v, b_w, b_a, b_g,
             b_kk, b_y, b_mid, b_z, b_at, b_mixs, b_x, b_vf, out + O_VF, true, stream);
  ln_kernel<<<B_ * M_, 256, 0, stream>>>(b_x, nullptr, P(95), P(94), b_x, B_ * M_);

  // ---- 3) RWKV block 1 (ctx = ego, Lc = 1) + LN n2 ----
  rwkv_block(d_in, ATTN1, ego, 1, b_seq, b_h, b_mx, b_r, b_k, b_v, b_w, b_a, b_g,
             b_kk, b_y, b_mid, b_z, b_at, b_mixs, b_x, b_vf, out + O_VF, false, stream);
  ln_kernel<<<B_ * M_, 256, 0, stream>>>(b_x, nullptr, P(97), P(96), b_x, B_ * M_);

  // ---- 4) FFN + LN n3 ----
  gemm(b_x,   P(90), P(88), b_ffn, B_ * M_, C_, FFN_, ACT_RELU, stream);
  gemm(b_ffn, P(91), P(89), b_x2,  B_ * M_, FFN_, C_, ACT_NONE, stream);
  ln_kernel<<<B_ * M_, 256, 0, stream>>>(b_x2, nullptr, P(99), P(98), b_x, B_ * M_);

  // ---- 5) AdaLN modulation from time embedding (mish -> mod_w) ----
  mish_kernel<<<ewg(B_ * C_), 256, 0, stream>>>(te, b_te, B_ * C_);
  gemm(b_te, P(93), P(92), b_ss, B_, C_, 2 * C_, ACT_NONE, stream);
  modulate_kernel<<<ewg(B_ * M_ * C_), 256, 0, stream>>>(b_x, b_ss);

  // ---- 6) regression branch ----
  gemm(b_x,   P(103), P(100), b_hh,  B_ * M_, C_, C_, ACT_RELU, stream);
  gemm(b_hh,  P(104), P(101), b_hh2, B_ * M_, C_, C_, ACT_RELU, stream);
  gemm(b_hh2, P(105), P(102), b_reg, B_ * M_, C_, T_ * 3, ACT_NONE, stream);
  poses_kernel<<<ewg(B_ * M_ * T_), 256, 0, stream>>>(b_reg, pts, out + O_POSES);

  // ---- 7) heads on cat(x, delta) ----
  cat_kernel<<<ewg(B_ * M_ * DIN_), 256, 0, stream>>>(b_x, b_reg, b_cat);
  run_head(b_cat, d_in, SCORE0, out + O_SCORE, 6,          b_hh, b_hh2, stream);
  run_head(b_cat, d_in, COL0,   out + O_COL,   2 * 40 * 9, b_hh, b_hh2, stream);
  run_head(b_cat, d_in, AREA0,  out + O_AREA,  40 * 2,     b_hh, b_hh2, stream);
  // v_first already written to out + O_VF by block 0.
}
